// SAGEResBlock_4329327034526
// MI455X (gfx1250) — matrix operand order repaired and checked by reading, not compiled, not run on hardware
//
#include <hip/hip_runtime.h>
#include <hip/hip_bf16.h>

#define D_IN  128
#define D_OUT 256
#define BN_EPS 1e-5f

typedef __attribute__((ext_vector_type(16))) _Float16 v16h;
typedef __attribute__((ext_vector_type(8)))  _Float16 v8h;
typedef __attribute__((ext_vector_type(8)))  float    v8f;

// ---------------- zero fill ----------------
__global__ void sage_zero_f32(float* __restrict__ p, int n) {
    int i = blockIdx.x * blockDim.x + threadIdx.x;
    int stride = gridDim.x * blockDim.x;
    for (; i < n; i += stride) p[i] = 0.0f;
}

// ---------------- scatter-mean accumulation ----------------
// one thread per (edge, 4-feature chunk): 32 threads per edge
__global__ void sage_scatter(const float* __restrict__ x,
                             const int*   __restrict__ ei,   // [2*E], row0=src row1=dst
                             float* __restrict__ agg,        // [N*D_IN] accumulators
                             float* __restrict__ cnt,        // [N]
                             int E) {
    int tid = blockIdx.x * blockDim.x + threadIdx.x;
    if (tid >= E * 32) return;
    int e = tid >> 5;
    int f = (tid & 31) << 2;
    int src = ei[e];
    int dst = ei[E + e];
    const float4 v = *(const float4*)(x + (size_t)src * D_IN + f);
    float* a = agg + (size_t)dst * D_IN + f;
    atomicAdd(a + 0, v.x);
    atomicAdd(a + 1, v.y);
    atomicAdd(a + 2, v.z);
    atomicAdd(a + 3, v.w);
    if (f == 0) atomicAdd(cnt + dst, 1.0f);
}

// ---------------- stage x and agg/cnt as f16 ----------------
__global__ void sage_prep_f16(const float* __restrict__ x,
                              const float* __restrict__ agg,
                              const float* __restrict__ cnt,
                              _Float16* __restrict__ xh,
                              _Float16* __restrict__ aggh,
                              int total) {                     // N * D_IN
    int i = blockIdx.x * blockDim.x + threadIdx.x;
    if (i >= total) return;
    const int row = i >> 7;                                    // D_IN == 128
    const float inv = 1.0f / fmaxf(cnt[row], 1.0f);
    xh[i]   = (_Float16)x[i];
    aggh[i] = (_Float16)(agg[i] * inv);
}

// ---------------- stage weights as f16 (once, not per m-block) ----------------
__global__ void sage_wcvt_f16(const float* __restrict__ Wl,
                              const float* __restrict__ Wr,
                              const float* __restrict__ Ws,
                              _Float16* __restrict__ Wlh,
                              _Float16* __restrict__ Wrh,
                              _Float16* __restrict__ Wsh,
                              int total) {                     // D_OUT * D_IN
    int i = blockIdx.x * blockDim.x + threadIdx.x;
    if (i >= total) return;
    Wlh[i] = (_Float16)Wl[i];
    Wrh[i] = (_Float16)Wr[i];
    Wsh[i] = (_Float16)Ws[i];
}

// ---------------- fragment loaders (pure f16 vector loads) ----------------
// A-matrix 16x32 layout: element j -> K = kk + half*16 + j (contiguous, matches B)
__device__ __forceinline__ v16h loadA16(const _Float16* __restrict__ row,
                                        int kk, int half) {
    return *(const v16h*)(row + kk + half * 16);
}

// B-matrix 32x16 layout: lane-half h holds K = kk+h*16 .. kk+h*16+15 (contiguous)
__device__ __forceinline__ v16h loadB16(const _Float16* __restrict__ wrow,
                                        int kk, int half) {
    return *(const v16h*)(wrow + kk + half * 16);
}

// ---------------- fused triple GEMM + bias + BN partial stats ----------------
// block = 128 threads (4 waves). block b covers rows [b*16, b*16+16);
// wave w covers output columns [w*64, w*64+64) as 4 WMMA tiles.
__global__ __launch_bounds__(128) void sage_gemm_wmma(
    const _Float16* __restrict__ xh,    // [N, 128] f16
    const _Float16* __restrict__ aggh,  // [N, 128] f16 (already mean-normalized)
    const _Float16* __restrict__ Wlh,   // [256, 128] f16
    const float*    __restrict__ bl,    // [256]
    const _Float16* __restrict__ Wrh,   // [256, 128] f16
    const _Float16* __restrict__ Wsh,   // [256, 128] f16
    const float*    __restrict__ bres,  // [256]
    float* __restrict__ out_pre,        // [N, 256]  (pre-BN, lives in d_out)
    float* __restrict__ ident,          // [N, 256]
    float* __restrict__ bn_sum,         // [256]
    float* __restrict__ bn_sumsq) {     // [256]
    const int lane = threadIdx.x & 31;
    const int wave = threadIdx.x >> 5;
    const int half = lane >> 4;
    const int l16  = lane & 15;
    const int m_base = blockIdx.x * 16;
    const int o_base = wave * 64;

    const int mrow = m_base + l16;                 // the A-row this lane feeds
    const _Float16* xrow = xh   + (size_t)mrow * D_IN;
    const _Float16* arow = aggh + (size_t)mrow * D_IN;

    v8f acc_out[4] = {};   // agg@Wl^T + x@Wr^T
    v8f acc_id[4]  = {};   // x@Wres^T

#pragma unroll
    for (int kk = 0; kk < D_IN; kk += 32) {
        const v16h a_agg = loadA16(arow, kk, half);
        const v16h a_x   = loadA16(xrow, kk, half);
#pragma unroll
        for (int t = 0; t < 4; ++t) {
            const size_t orow = (size_t)(o_base + t * 16 + l16) * D_IN;
            const v16h b_l = loadB16(Wlh + orow, kk, half);
            const v16h b_r = loadB16(Wrh + orow, kk, half);
            const v16h b_s = loadB16(Wsh + orow, kk, half);
            acc_out[t] = __builtin_amdgcn_wmma_f32_16x16x32_f16(
                false, a_agg, false, b_l, (short)0, acc_out[t], false, false);
            acc_out[t] = __builtin_amdgcn_wmma_f32_16x16x32_f16(
                false, a_x,   false, b_r, (short)0, acc_out[t], false, false);
            acc_id[t]  = __builtin_amdgcn_wmma_f32_16x16x32_f16(
                false, a_x,   false, b_s, (short)0, acc_id[t],  false, false);
        }
    }

    // Epilogue: C/D layout -> lane element r is (m = m_base + half*8 + r, n)
#pragma unroll
    for (int t = 0; t < 4; ++t) {
        const int   n  = o_base + t * 16 + l16;
        const float bO = bl[n];
        const float bI = bres[n];
        float s = 0.0f, s2 = 0.0f;
#pragma unroll
        for (int r = 0; r < 8; ++r) {
            const int m = m_base + half * 8 + r;
            const float v  = acc_out[t][r] + bO;
            const float iv = acc_id[t][r]  + bI;
            out_pre[(size_t)m * D_OUT + n] = v;
            ident[(size_t)m * D_OUT + n]   = iv;
            s  += v;
            s2 += v * v;
        }
        atomicAdd(&bn_sum[n],   s);
        atomicAdd(&bn_sumsq[n], s2);
    }
}

// ---------------- BN finalize + ReLU + residual ----------------
__global__ void sage_bn_final(const float* __restrict__ ident,
                              const float* __restrict__ bn_sum,
                              const float* __restrict__ bn_sumsq,
                              const float* __restrict__ gamma,
                              const float* __restrict__ beta,
                              float* __restrict__ out,   // holds out_pre on entry
                              float inv_n, int total) {
    int idx = blockIdx.x * blockDim.x + threadIdx.x;
    if (idx >= total) return;
    const int n = idx & (D_OUT - 1);
    const float mean  = bn_sum[n] * inv_n;
    const float var   = bn_sumsq[n] * inv_n - mean * mean;
    const float scale = rsqrtf(var + BN_EPS) * gamma[n];
    float v = (out[idx] - mean) * scale + beta[n];
    v = fmaxf(v, 0.0f);
    out[idx] = v + ident[idx];
}

extern "C" void kernel_launch(void* const* d_in, const int* in_sizes, int n_in,
                              void* d_out, int out_size, void* d_ws, size_t ws_size,
                              hipStream_t stream) {
    const float* x     = (const float*)d_in[0];
    const int*   ei    = (const int*)  d_in[1];
    const float* Wl    = (const float*)d_in[2];
    const float* bl    = (const float*)d_in[3];
    const float* Wr    = (const float*)d_in[4];
    const float* Wres  = (const float*)d_in[5];
    const float* bres  = (const float*)d_in[6];
    const float* gamma = (const float*)d_in[7];
    const float* beta  = (const float*)d_in[8];

    const int N = in_sizes[0] / D_IN;      // 50000
    const int E = in_sizes[1] / 2;         // 640000

    // workspace layout (all offsets 32B-aligned)
    char* ws = (char*)d_ws;
    size_t off = 0;
    float* agg      = (float*)(ws + off); off += (size_t)N * D_IN * 4;   // 25.6 MB
    float* cnt      = (float*)(ws + off); off += (size_t)N * 4;          // 0.2 MB (contiguous after agg)
    float* ident    = (float*)(ws + off); off += (size_t)N * D_OUT * 4;  // 51.2 MB
    float* bn_sum   = (float*)(ws + off); off += (size_t)2 * D_OUT * 4;  // 2 KB (sum + sumsq)
    float* bn_sumsq = bn_sum + D_OUT;
    off = (off + 31) & ~(size_t)31;
    _Float16* xh   = (_Float16*)(ws + off); off += (size_t)N * D_IN * 2; // 12.8 MB
    _Float16* aggh = (_Float16*)(ws + off); off += (size_t)N * D_IN * 2; // 12.8 MB
    _Float16* Wlh  = (_Float16*)(ws + off); off += (size_t)D_OUT * D_IN * 2;
    _Float16* Wrh  = (_Float16*)(ws + off); off += (size_t)D_OUT * D_IN * 2;
    _Float16* Wsh  = (_Float16*)(ws + off); off += (size_t)D_OUT * D_IN * 2;

    float* out_pre = (float*)d_out;        // pre-BN tensor staged in d_out

    // 1) zero accumulators (agg + cnt contiguous) and BN stats
    {
        int nz = N * D_IN + N;
        sage_zero_f32<<<(nz + 255) / 256, 256, 0, stream>>>(agg, nz);
        sage_zero_f32<<<2, 256, 0, stream>>>(bn_sum, 2 * D_OUT);
    }
    // 2) scatter-mean accumulation
    {
        int nt = E * 32;
        sage_scatter<<<(nt + 255) / 256, 256, 0, stream>>>(x, ei, agg, cnt, E);
    }
    // 3) stage f16 operands (x, agg/cnt, weights)
    {
        int total = N * D_IN;
        sage_prep_f16<<<(total + 255) / 256, 256, 0, stream>>>(x, agg, cnt, xh, aggh, total);
        int wt = D_OUT * D_IN;
        sage_wcvt_f16<<<(wt + 255) / 256, 256, 0, stream>>>(Wl, Wr, Wres, Wlh, Wrh, Wsh, wt);
    }
    // 4) fused WMMA triple-GEMM + biases + BN partial sums
    sage_gemm_wmma<<<N / 16, 128, 0, stream>>>(xh, aggh, Wlh, bl, Wrh, Wsh, bres,
                                               out_pre, ident, bn_sum, bn_sumsq);
    // 5) BN finalize + ReLU + residual add
    {
        int total = N * D_OUT;
        sage_bn_final<<<(total + 255) / 256, 256, 0, stream>>>(
            ident, bn_sum, bn_sumsq, gamma, beta, out_pre, 1.0f / (float)N, total);
    }
}
